// DGCNN_4681514353032
// MI455X (gfx1250) — matrix operand order repaired and checked by reading, not compiled, not run on hardware
//
#include <hip/hip_runtime.h>

#define BATCH 8
#define NPTS  4096
#define KNN   20

typedef __attribute__((ext_vector_type(16))) _Float16 v16h;
typedef __attribute__((ext_vector_type(8)))  float    v8f;

union V16H { v16h v; float4 f4[2]; };

__device__ __forceinline__ float lrelu02(float v) { return fmaxf(v, 0.2f * v); }

// ---------------------------------------------------------------------------
// KNN: one thread per query point, candidates staged through LDS in tiles of
// 128 points. Sorted top-20 (ascending distance) kept in registers.
// ---------------------------------------------------------------------------
template <int D>
__global__ __launch_bounds__(128) void knn_kernel(const float* __restrict__ X,
                                                  int ldX,
                                                  int* __restrict__ idx) {
  __shared__ float tile[128 * D];
  const int b = blockIdx.y;
  const int n = blockIdx.x * 128 + threadIdx.x;
  const float* Xb = X + (size_t)b * NPTS * ldX;

  float q[D];
#pragma unroll
  for (int d = 0; d < D; ++d) q[d] = Xb[(size_t)n * ldX + d];

  float bd[KNN];
  int bi[KNN];
#pragma unroll
  for (int t = 0; t < KNN; ++t) { bd[t] = 3.4e38f; bi[t] = 0; }

  for (int t0 = 0; t0 < NPTS; t0 += 128) {
    __syncthreads();
    for (int e = threadIdx.x; e < 128 * D; e += 128) {
      int c = e / D, d = e - c * D;
      tile[e] = Xb[(size_t)(t0 + c) * ldX + d];
    }
    __syncthreads();
    for (int j = 0; j < 128; ++j) {
      float dist = 0.f;
#pragma unroll
      for (int d = 0; d < D; ++d) {
        float df = q[d] - tile[j * D + d];
        dist = fmaf(df, df, dist);
      }
      if (dist < bd[KNN - 1]) {  // rare path: bubble-insert keeps list sorted
        float cd = dist;
        int ci = t0 + j;
#pragma unroll
        for (int t = 0; t < KNN; ++t) {
          if (cd < bd[t]) {
            float tf = bd[t]; bd[t] = cd; cd = tf;
            int   ti = bi[t]; bi[t] = ci; ci = ti;
          }
        }
      }
    }
  }
  int* op = idx + ((size_t)b * NPTS + n) * KNN;
#pragma unroll
  for (int t = 0; t < KNN; ++t) op[t] = bi[t];
}

// ---------------------------------------------------------------------------
// f32 -> f16 staging with zero K-padding (rows = B*N flat, row-major output)
// ---------------------------------------------------------------------------
__global__ void cvt_f16_kernel(const float* __restrict__ in,
                               _Float16* __restrict__ out, int total, int ldin,
                               int Din, int Dp) {
  int gid = blockIdx.x * blockDim.x + threadIdx.x;
  if (gid >= total) return;
  int dp = gid % Dp;
  int r = gid / Dp;
  float v = (dp < Din) ? in[(size_t)r * ldin + dp] : 0.0f;
  out[gid] = (_Float16)v;
}

// ---------------------------------------------------------------------------
// Weight pre-swizzle into CDNA5 WMMA B-matrix (16-bit, 16x16x32) lane layout:
// flat = ((kt*numOt + ot)*32 + lane)*16 + i ;  k = kt*32 + (lane>>4)*8 + (i&7) + (i>>3)*16,
// col = ot*16 + (lane&15).   Edge mode stacks [Wa ; Wb - Wa] (2O columns).
// ---------------------------------------------------------------------------
__global__ void prep_w_edge(const float* __restrict__ W,
                            _Float16* __restrict__ Wh, int total, int D,
                            int O) {
  int gid = blockIdx.x * blockDim.x + threadIdx.x;
  if (gid >= total) return;
  int O2 = 2 * O;
  int numOt = O2 >> 4;
  int i = gid & 15;
  int lane = (gid >> 4) & 31;
  int rest = gid >> 9;
  int ot = rest % numOt;
  int kt = rest / numOt;
  int k = kt * 32 + (lane >> 4) * 8 + (i & 7) + (i >> 3) * 16;
  int oc = ot * 16 + (lane & 15);
  float v = 0.f;
  if (k < D) {
    if (oc < O) v = W[(size_t)oc * (2 * D) + k];                       // Wa
    else {
      int o = oc - O;
      v = W[(size_t)o * (2 * D) + D + k] - W[(size_t)o * (2 * D) + k]; // Wb-Wa
    }
  }
  Wh[gid] = (_Float16)v;
}

__global__ void prep_w_plain(const float* __restrict__ W,
                             _Float16* __restrict__ Wh, int total, int K,
                             int O) {
  int gid = blockIdx.x * blockDim.x + threadIdx.x;
  if (gid >= total) return;
  int numOt = O >> 4;
  int i = gid & 15;
  int lane = (gid >> 4) & 31;
  int rest = gid >> 9;
  int ot = rest % numOt;
  int kt = rest / numOt;
  int k = kt * 32 + (lane >> 4) * 8 + (i & 7) + (i >> 3) * 16;
  int o = ot * 16 + (lane & 15);
  Wh[gid] = (_Float16)W[(size_t)o * K + k];  // Bmat[k][o] = W^T
}

// ---------------------------------------------------------------------------
// Dense GEMM:  Y[r, oc] = sum_k Xh[r,k] * Bmat[k,oc]
// 1 wave -> 16 rows x 32 cols, f16 WMMA 16x16x32, f32 accumulate.
// ---------------------------------------------------------------------------
__global__ __launch_bounds__(32) void gemm_f16_wmma(
    const _Float16* __restrict__ Xh, const _Float16* __restrict__ Wh,
    float* __restrict__ Y, int Dp, int O2) {
  const int lane = threadIdx.x;
  const int row0 = blockIdx.x * 16;
  const int o0 = blockIdx.y * 32;
  const int half = lane >> 4;
  const int m = lane & 15;
  const int numOt16 = O2 >> 4;
  const int ksteps = Dp >> 5;

  v8f acc0 = {};
  v8f acc1 = {};
  const _Float16* arow = Xh + (size_t)(row0 + m) * Dp + half * 8;
  for (int ks = 0; ks < ksteps; ++ks) {
    V16H a;
    a.f4[0] = *(const float4*)(arow + ks * 32);
    a.f4[1] = *(const float4*)(arow + ks * 32 + 16);
    if (ks + 1 < ksteps)
      __builtin_prefetch(arow + (ks + 1) * 32, 0, 3);  // WGP-scope prefetch
    const _Float16* wp0 =
        Wh + (((size_t)ks * numOt16 + (blockIdx.y * 2 + 0)) * 32 + lane) * 16;
    const _Float16* wp1 =
        Wh + (((size_t)ks * numOt16 + (blockIdx.y * 2 + 1)) * 32 + lane) * 16;
    V16H b0, b1;
    b0.f4[0] = *(const float4*)(wp0);
    b0.f4[1] = *(const float4*)(wp0 + 8);
    b1.f4[0] = *(const float4*)(wp1);
    b1.f4[1] = *(const float4*)(wp1 + 8);
    acc0 = __builtin_amdgcn_wmma_f32_16x16x32_f16(false, a.v, false, b0.v,
                                                  (short)0, acc0, false, false);
    acc1 = __builtin_amdgcn_wmma_f32_16x16x32_f16(false, a.v, false, b1.v,
                                                  (short)0, acc1, false, false);
  }
  const int ncol = lane & 15;
#pragma unroll
  for (int v = 0; v < 8; ++v) {
    int r = row0 + v + half * 8;
    Y[(size_t)r * O2 + o0 + ncol] = acc0[v];
    Y[(size_t)r * O2 + o0 + 16 + ncol] = acc1[v];
  }
}

// ---------------------------------------------------------------------------
// Edge-conv epilogue: h_j = A[idx_j] + C[n]; BN + LeakyReLU + max over k.
// AC layout: [.., 0:O) = A, [O:2O) = C. Writes strided into xc concat buffer.
// ---------------------------------------------------------------------------
__global__ void edge_max_kernel(const float* __restrict__ AC,
                                const int* __restrict__ idx,
                                const float* __restrict__ g,
                                const float* __restrict__ bb,
                                float* __restrict__ xc, int O, int off) {
  int gid = blockIdx.x * blockDim.x + threadIdx.x;
  int total = BATCH * NPTS * O;
  if (gid >= total) return;
  int o = gid % O;
  int n = (gid / O) % NPTS;
  int b = gid / (O * NPTS);
  int O2 = 2 * O;
  const float* ACb = AC + (size_t)b * NPTS * O2;
  float c = ACb[(size_t)n * O2 + O + o];
  float rs = rsqrtf(1.0f + 1e-5f);
  float s = g[o] * rs, bi = bb[o];
  const int* ip = idx + ((size_t)b * NPTS + n) * KNN;
  float m = -3.4e38f;
#pragma unroll 4
  for (int j = 0; j < KNN; ++j) {
    int nb = ip[j];
    float a = ACb[(size_t)nb * O2 + o];
    m = fmaxf(m, lrelu02((a + c) * s + bi));
  }
  xc[((size_t)b * NPTS + n) * 320 + off + o] = m;
}

// ---------------------------------------------------------------------------
// W5 layer fused with global pooling: per wave, a 32-wide output slice is
// accumulated over all N row-tiles; BN+LeakyReLU in-register, running max/sum
// reduced across half-waves with shfl_xor(16). Never materializes h.
// ---------------------------------------------------------------------------
__global__ __launch_bounds__(32) void gemm_pool_wmma(
    const _Float16* __restrict__ Xh, const _Float16* __restrict__ Wh,
    const float* __restrict__ g, const float* __restrict__ bia,
    float* __restrict__ gfeat, int Dp, int O2) {
  const int lane = threadIdx.x;
  const int b = blockIdx.x;
  const int o0 = blockIdx.y * 32;
  const int half = lane >> 4;
  const int m = lane & 15;
  const int numOt16 = O2 >> 4;
  const int ksteps = Dp >> 5;

  float rs = rsqrtf(1.0f + 1e-5f);
  int oA = o0 + (lane & 15), oB = oA + 16;
  float sA = g[oA] * rs, bA = bia[oA];
  float sB = g[oB] * rs, bB = bia[oB];
  float mx0 = -3.4e38f, mx1 = -3.4e38f, sm0 = 0.f, sm1 = 0.f;

  const _Float16* Xb = Xh + (size_t)b * NPTS * Dp;
  for (int nt = 0; nt < NPTS / 16; ++nt) {
    v8f acc0 = {};
    v8f acc1 = {};
    const _Float16* arow = Xb + (size_t)(nt * 16 + m) * Dp + half * 8;
    for (int ks = 0; ks < ksteps; ++ks) {
      V16H a;
      a.f4[0] = *(const float4*)(arow + ks * 32);
      a.f4[1] = *(const float4*)(arow + ks * 32 + 16);
      if (ks + 1 < ksteps)
        __builtin_prefetch(arow + (ks + 1) * 32, 0, 3);  // WGP-scope prefetch
      const _Float16* wp0 =
          Wh + (((size_t)ks * numOt16 + (blockIdx.y * 2 + 0)) * 32 + lane) * 16;
      const _Float16* wp1 =
          Wh + (((size_t)ks * numOt16 + (blockIdx.y * 2 + 1)) * 32 + lane) * 16;
      V16H b0, b1;
      b0.f4[0] = *(const float4*)(wp0);
      b0.f4[1] = *(const float4*)(wp0 + 8);
      b1.f4[0] = *(const float4*)(wp1);
      b1.f4[1] = *(const float4*)(wp1 + 8);
      acc0 = __builtin_amdgcn_wmma_f32_16x16x32_f16(false, a.v, false, b0.v,
                                                    (short)0, acc0, false,
                                                    false);
      acc1 = __builtin_amdgcn_wmma_f32_16x16x32_f16(false, a.v, false, b1.v,
                                                    (short)0, acc1, false,
                                                    false);
    }
#pragma unroll
    for (int v = 0; v < 8; ++v) {
      float h0 = lrelu02(acc0[v] * sA + bA);
      float h1 = lrelu02(acc1[v] * sB + bB);
      mx0 = fmaxf(mx0, h0); sm0 += h0;
      mx1 = fmaxf(mx1, h1); sm1 += h1;
    }
  }
  // lanes L and L^16 hold the same output channel over disjoint row halves
  mx0 = fmaxf(mx0, __shfl_xor(mx0, 16, 32));
  sm0 += __shfl_xor(sm0, 16, 32);
  mx1 = fmaxf(mx1, __shfl_xor(mx1, 16, 32));
  sm1 += __shfl_xor(sm1, 16, 32);
  if (lane < 16) {
    float invN = 1.0f / (float)NPTS;
    float* gf = gfeat + (size_t)b * 2048;
    gf[oA] = mx0;
    gf[1024 + oA] = sm0 * invN;
    gf[oB] = mx1;
    gf[1024 + oB] = sm1 * invN;
  }
}

// ---------------------------------------------------------------------------
// FC head (tiny) + broadcast of logits to [B, N, 5].
// ---------------------------------------------------------------------------
__global__ __launch_bounds__(256) void fc_kernel(
    const float* __restrict__ gfeat, const float* __restrict__ Wl1,
    const float* __restrict__ g6, const float* __restrict__ b6,
    const float* __restrict__ Wl2, const float* __restrict__ bl2,
    const float* __restrict__ g7, const float* __restrict__ b7,
    const float* __restrict__ Wl3, const float* __restrict__ bl3,
    float* __restrict__ out) {
  __shared__ float sg[2048];
  __shared__ float s1[512];
  __shared__ float s2[256];
  __shared__ float s3[8];
  int b = blockIdx.x, t = threadIdx.x;
  const float* gf = gfeat + (size_t)b * 2048;
  for (int i = t; i < 2048; i += 256) sg[i] = gf[i];
  __syncthreads();
  float rs = rsqrtf(1.0f + 1e-5f);
  for (int o = t; o < 512; o += 256) {
    const float* w = Wl1 + (size_t)o * 2048;
    float acc = 0.f;
    for (int k = 0; k < 2048; ++k) acc = fmaf(sg[k], w[k], acc);
    s1[o] = lrelu02(acc * g6[o] * rs + b6[o]);
  }
  __syncthreads();
  {
    const float* w = Wl2 + (size_t)t * 512;
    float acc = 0.f;
    for (int k = 0; k < 512; ++k) acc = fmaf(s1[k], w[k], acc);
    acc += bl2[t];
    s2[t] = lrelu02(acc * g7[t] * rs + b7[t]);
  }
  __syncthreads();
  if (t < 5) {
    const float* w = Wl3 + (size_t)t * 256;
    float acc = 0.f;
    for (int k = 0; k < 256; ++k) acc = fmaf(s2[k], w[k], acc);
    s3[t] = acc + bl3[t];
  }
  __syncthreads();
  float* ob = out + (size_t)b * NPTS * 5;
  for (int p = t; p < NPTS * 5; p += 256) ob[p] = s3[p % 5];
}

// ---------------------------------------------------------------------------
extern "C" void kernel_launch(void* const* d_in, const int* in_sizes, int n_in,
                              void* d_out, int out_size, void* d_ws,
                              size_t ws_size, hipStream_t stream) {
  (void)in_sizes; (void)n_in; (void)out_size; (void)ws_size;
  const float* xyz = (const float*)d_in[0];
  const float* W1 = (const float*)d_in[1];
  const float* g1 = (const float*)d_in[2];
  const float* b1 = (const float*)d_in[3];
  const float* W2 = (const float*)d_in[4];
  const float* g2 = (const float*)d_in[5];
  const float* b2 = (const float*)d_in[6];
  const float* W3 = (const float*)d_in[7];
  const float* g3 = (const float*)d_in[8];
  const float* b3 = (const float*)d_in[9];
  const float* W4 = (const float*)d_in[10];
  const float* g4 = (const float*)d_in[11];
  const float* b4 = (const float*)d_in[12];
  const float* W5 = (const float*)d_in[13];
  const float* g5 = (const float*)d_in[14];
  const float* b5 = (const float*)d_in[15];
  const float* Wl1 = (const float*)d_in[16];
  const float* g6 = (const float*)d_in[17];
  const float* b6 = (const float*)d_in[18];
  const float* Wl2 = (const float*)d_in[19];
  const float* bl2 = (const float*)d_in[20];
  const float* g7 = (const float*)d_in[21];
  const float* b7 = (const float*)d_in[22];
  const float* Wl3 = (const float*)d_in[23];
  const float* bl3 = (const float*)d_in[24];

  char* wsb = (char*)d_ws;
  size_t off = 0;
  auto take = [&](size_t bytes) -> char* {
    char* p = wsb + off;
    off = (off + bytes + 255) & ~(size_t)255;
    return p;
  };
  float* xcp = (float*)take((size_t)BATCH * NPTS * 320 * 4);        // 42 MB
  float* ACp = (float*)take((size_t)BATCH * NPTS * 256 * 4);        // 34 MB
  _Float16* Xhp = (_Float16*)take((size_t)BATCH * NPTS * 320 * 2);  // 21 MB
  _Float16* Whp = (_Float16*)take((size_t)320 * 1024 * 2);          // 0.7 MB
  int* idxp = (int*)take((size_t)BATCH * NPTS * KNN * 4);           // 2.6 MB
  float* gfeatp = (float*)take((size_t)BATCH * 2048 * 4);

  const int rows = BATCH * NPTS;

  auto edge_layer = [&](const float* Xin, int ldX, int Din, int Dp, int O,
                        const float* W, const float* gg, const float* bb,
                        int coff, bool d3) {
    if (d3)
      knn_kernel<3><<<dim3(NPTS / 128, BATCH), 128, 0, stream>>>(Xin, ldX,
                                                                 idxp);
    else
      knn_kernel<64><<<dim3(NPTS / 128, BATCH), 128, 0, stream>>>(Xin, ldX,
                                                                  idxp);
    int tot = rows * Dp;
    cvt_f16_kernel<<<(tot + 255) / 256, 256, 0, stream>>>(Xin, Xhp, tot, ldX,
                                                          Din, Dp);
    int O2 = 2 * O;
    int wt = Dp * O2;
    prep_w_edge<<<(wt + 255) / 256, 256, 0, stream>>>(W, Whp, wt, Din, O);
    gemm_f16_wmma<<<dim3(rows / 16, O2 / 32), 32, 0, stream>>>(Xhp, Whp, ACp,
                                                               Dp, O2);
    int et = rows * O;
    edge_max_kernel<<<(et + 255) / 256, 256, 0, stream>>>(ACp, idxp, gg, bb,
                                                          xcp, O, coff);
  };

  edge_layer(xyz,        3,   3,  32,  64, W1, g1, b1,   0, true);
  edge_layer(xcp + 0,   320, 64,  64,  64, W2, g2, b2,  64, false);
  edge_layer(xcp + 64,  320, 64,  64,  64, W3, g3, b3, 128, false);
  edge_layer(xcp + 128, 320, 64,  64, 128, W4, g4, b4, 192, false);

  // W5 layer fused with global max/mean pooling
  {
    int tot = rows * 320;
    cvt_f16_kernel<<<(tot + 255) / 256, 256, 0, stream>>>(xcp, Xhp, tot, 320,
                                                          320, 320);
    int wt = 320 * 1024;
    prep_w_plain<<<(wt + 255) / 256, 256, 0, stream>>>(W5, Whp, wt, 320, 1024);
    gemm_pool_wmma<<<dim3(BATCH, 1024 / 32), 32, 0, stream>>>(Xhp, Whp, g5, b5,
                                                              gfeatp, 320,
                                                              1024);
  }

  fc_kernel<<<BATCH, 256, 0, stream>>>(gfeatp, Wl1, g6, b6, Wl2, bl2, g7, b7,
                                       Wl3, bl3, (float*)d_out);
}
